// HeroesCell_82695300317805
// MI455X (gfx1250) — matrix operand order repaired and checked by reading, not compiled, not run on hardware
//
#include <hip/hip_runtime.h>

typedef __attribute__((ext_vector_type(16))) __bf16 v16bf;
typedef __attribute__((ext_vector_type(8)))  __bf16 v8bf;
typedef __attribute__((ext_vector_type(8)))  float  v8f;

#define B_      32768
#define U_      256
#define ROWS    64    // rows per block, 4 m-tiles of 16
#define MT      4
#define THREADS 256
#define LDSTR   264   // bf16 elems per LDS row (256 + pad), 528B (16B aligned rows)
#define A0STR   136   // 272B rows
#define A1STR   72    // 144B rows
#define PKU     65536 // bf16 elems per packed 256x256 matrix

// packed-weight offsets in d_ws (bf16 element units)
#define OFF_SCHAT_C 0
#define OFF_SCHAT_V (1*PKU)
#define OFF_SVHAT_V (2*PKU)
#define OFF_SVHAT_C (3*PKU)
#define OFF_WXC     (4*PKU)
#define OFF_WHC     (8*PKU)
#define OFF_WXV     (12*PKU)
#define OFF_WHV     (16*PKU)
#define OFF_WPC0    (20*PKU)
#define OFF_WPV0    (20*PKU + 32768)
#define OFF_WPC1    (20*PKU + 65536)
#define OFF_WPV1    (20*PKU + 65536 + 8192)
// total = 1,392,640 bf16 elems = 2,785,280 bytes of d_ws

// output offsets (floats): (h_c_p, h_v_p, h_c_n, h_v_n, s_c_n, s_v_n, g_new)
#define BU_    ((size_t)B_ * U_)
#define O_HCP  ((size_t)0)
#define O_HVP  ((size_t)B_)
#define O_HCN  ((size_t)2 * B_)
#define O_HVN  (O_HCN + BU_)
#define O_SCN  (O_HVN + BU_)
#define O_SVN  (O_SCN + BU_)
#define O_GNEW (O_SVN + BU_)

// ---------- bf16 <-> f32 helpers (RNE) ---------------------------------------
__device__ __forceinline__ __bf16 f2bf(float f) {
    union { float f; unsigned u; } a; a.f = f;
    unsigned r = a.u + 0x7FFFu + ((a.u >> 16) & 1u);
    unsigned short h = (unsigned short)(r >> 16);
    __bf16 b; __builtin_memcpy(&b, &h, 2); return b;
}
__device__ __forceinline__ float bf2f(__bf16 b) {
    unsigned short h; __builtin_memcpy(&h, &b, 2);
    union { unsigned u; float f; } a; a.u = ((unsigned)h) << 16; return a.f;
}
__device__ __forceinline__ float sigmoidf_(float x) { return 1.f / (1.f + __expf(-x)); }
__device__ __forceinline__ float leakyf_(float x) { return x > 0.f ? x : 0.3f * x; }

// ---------- WMMA helpers -----------------------------------------------------
__device__ __forceinline__ v8f wmma_bf16(v16bf a, v16bf b, v8f c) {
    return __builtin_amdgcn_wmma_f32_16x16x32_bf16(false, a, false, b, (short)0, c, false, false);
}

// A fragment (16x32 bf16) from row-major LDS tile. ISA 7.12.2:
// lane l: row = l%16, K = {8g..8g+7, 16+8g..16+8g+7}, g = l/16
__device__ __forceinline__ v16bf afrag(const __bf16* s, int stride, int mrow, int k0, int lane) {
    int m = lane & 15, g = lane >> 4;
    const __bf16* p = s + (mrow + m) * stride + k0 + 8 * g;
    v8bf lo = *(const v8bf*)p;
    v8bf hi = *(const v8bf*)(p + 16);
    return __builtin_shufflevector(lo, hi, 0,1,2,3,4,5,6,7,8,9,10,11,12,13,14,15);
}

// B fragment from pre-packed global weights: tile (j=ntile, t=ktile) is 1KB,
// lane's 16 bf16 are contiguous at lane*16.
__device__ __forceinline__ v16bf bfrag(const __bf16* pkb, int KT, int j, int t, int lane) {
    return *(const v16bf*)(pkb + ((size_t)(j * KT + t) << 9) + (lane << 4));
}

// ---------- weight pack kernel ----------------------------------------------
// W[K][N] fp32 row-major -> bf16 B-fragment tiles. ISA B layout (32x16):
// lane l: col = l%16, K = 8*(l/16) + e for e<8, 16 + 8*(l/16) + (e-8) for e>=8
__global__ void pack_bfrag(const float* __restrict__ src, __bf16* __restrict__ dst,
                           int K, int N) {
    int tid = blockIdx.x * blockDim.x + threadIdx.x;
    int KT = K >> 5;
    int tiles = KT * (N >> 4);
    if (tid >= tiles * 32) return;
    int tile = tid >> 5, lane = tid & 31;
    int j = tile / KT, t = tile - j * KT;
    int n = (j << 4) + (lane & 15);
    int kb = (t << 5) + ((lane >> 4) << 3);
    __bf16* d = dst + ((size_t)tile << 9) + (lane << 4);
    #pragma unroll
    for (int e = 0; e < 16; ++e)
        d[e] = f2bf(src[(size_t)(kb + e + ((e >> 3) << 3)) * N + n]);
}

// ---------- fused HeroesCell kernel -----------------------------------------
__global__ __launch_bounds__(THREADS) void heroes_fused(
    const float* __restrict__ x, const float* __restrict__ click,
    const float* __restrict__ h_c, const float* __restrict__ h_v,
    const float* __restrict__ s_c, const float* __restrict__ s_v,
    const int* __restrict__ g_prev,
    const float* __restrict__ bx_c, const float* __restrict__ bx_v,
    const float* __restrict__ bpc0, const float* __restrict__ bpc1,
    const float* __restrict__ Wfcc, const float* __restrict__ bfcc,
    const float* __restrict__ bpv0, const float* __restrict__ bpv1,
    const float* __restrict__ Wfcv, const float* __restrict__ bfcv,
    const __bf16* __restrict__ pk, float* __restrict__ out) {

    __shared__ __bf16 sX [ROWS * LDSTR];  // x tile (alive all steps)
    __shared__ __bf16 sB1[ROWS * LDSTR];  // h_c -> h_c_n -> h_v_n
    __shared__ __bf16 sHv[ROWS * LDSTR];  // h_v
    __shared__ __bf16 sT [ROWS * LDSTR];  // s_c_hat -> s_v_hat
    __shared__ __bf16 sA0[ROWS * A0STR];  // MLP hidden0
    __shared__ __bf16 sA1[ROWS * A1STR];  // MLP hidden1
    __shared__ float  sP [ROWS];          // h_c_p

    const int tid = threadIdx.x;
    const int rowBase = blockIdx.x * ROWS;

    // ---- prologue: stage x, h_c, h_v as bf16 in LDS -------------------------
    for (int i = tid; i < ROWS * 64; i += THREADS) {   // 64 float4 per 256-wide row
        int r = i >> 6, c = (i & 63) << 2;
        size_t go = (size_t)(rowBase + r) * U_ + c;
        int lo = r * LDSTR + c;
        float4 vx = *(const float4*)(x + go);
        sX[lo+0]=f2bf(vx.x); sX[lo+1]=f2bf(vx.y); sX[lo+2]=f2bf(vx.z); sX[lo+3]=f2bf(vx.w);
        float4 vc = *(const float4*)(h_c + go);
        sB1[lo+0]=f2bf(vc.x); sB1[lo+1]=f2bf(vc.y); sB1[lo+2]=f2bf(vc.z); sB1[lo+3]=f2bf(vc.w);
        float4 vv = *(const float4*)(h_v + go);
        sHv[lo+0]=f2bf(vv.x); sHv[lo+1]=f2bf(vv.y); sHv[lo+2]=f2bf(vv.z); sHv[lo+3]=f2bf(vv.w);
    }
    __syncthreads();

    const int w = tid >> 5, lane = tid & 31;
    const int ln = lane & 15, lg = lane >> 4;

    // ---- step 1: s_c_hat = gp ? h_v@W_schat_v : h_c@W_schat_c --------------
    // Each wave: 2 n-tiles, all 4 m-tiles (B frags reused 4x per load).
    for (int jj = 0; jj < 2; ++jj) {
        int j = jj * 8 + w;
        v8f a1[MT] = {}, a2[MT] = {};
        #pragma unroll
        for (int t = 0; t < 8; ++t) {
            v16bf b1 = bfrag(pk + OFF_SCHAT_C, 8, j, t, lane);
            v16bf b2 = bfrag(pk + OFF_SCHAT_V, 8, j, t, lane);
            #pragma unroll
            for (int mt = 0; mt < MT; ++mt) {
                a1[mt] = wmma_bf16(afrag(sB1, LDSTR, mt*16, t*32, lane), b1, a1[mt]);
                a2[mt] = wmma_bf16(afrag(sHv, LDSTR, mt*16, t*32, lane), b2, a2[mt]);
            }
        }
        int n = (j << 4) + ln;
        #pragma unroll
        for (int mt = 0; mt < MT; ++mt) {
            #pragma unroll
            for (int v = 0; v < 8; ++v) {
                int rl = mt*16 + v + 8*lg;
                int r = rowBase + rl;
                float val = (g_prev[(size_t)r*U_ + n] > 0) ? a2[mt][v] : a1[mt][v];
                sT[rl * LDSTR + n] = f2bf(val);
            }
        }
    }
    __syncthreads();

    // ---- step 2: click-branch gates, s_c_n / h_c_n -------------------------
    for (int jj = 0; jj < 2; ++jj) {
        int j = jj * 8 + w;
        v8f af[MT] = {}, ai[MT] = {}, ao[MT] = {}, ag[MT] = {};
        #pragma unroll
        for (int t = 0; t < 8; ++t) {      // x @ Wx_c
            v16bf bF = bfrag(pk + OFF_WXC + 0*PKU, 8, j, t, lane);
            v16bf bI = bfrag(pk + OFF_WXC + 1*PKU, 8, j, t, lane);
            v16bf bO = bfrag(pk + OFF_WXC + 2*PKU, 8, j, t, lane);
            v16bf bG = bfrag(pk + OFF_WXC + 3*PKU, 8, j, t, lane);
            #pragma unroll
            for (int mt = 0; mt < MT; ++mt) {
                v16bf a = afrag(sX, LDSTR, mt*16, t*32, lane);
                af[mt] = wmma_bf16(a, bF, af[mt]);
                ai[mt] = wmma_bf16(a, bI, ai[mt]);
                ao[mt] = wmma_bf16(a, bO, ao[mt]);
                ag[mt] = wmma_bf16(a, bG, ag[mt]);
            }
        }
        #pragma unroll
        for (int t = 0; t < 8; ++t) {      // s_c_hat @ Wh_c
            v16bf bF = bfrag(pk + OFF_WHC + 0*PKU, 8, j, t, lane);
            v16bf bI = bfrag(pk + OFF_WHC + 1*PKU, 8, j, t, lane);
            v16bf bO = bfrag(pk + OFF_WHC + 2*PKU, 8, j, t, lane);
            v16bf bG = bfrag(pk + OFF_WHC + 3*PKU, 8, j, t, lane);
            #pragma unroll
            for (int mt = 0; mt < MT; ++mt) {
                v16bf a = afrag(sT, LDSTR, mt*16, t*32, lane);
                af[mt] = wmma_bf16(a, bF, af[mt]);
                ai[mt] = wmma_bf16(a, bI, ai[mt]);
                ao[mt] = wmma_bf16(a, bO, ao[mt]);
                ag[mt] = wmma_bf16(a, bG, ag[mt]);
            }
        }
        int n = (j << 4) + ln;
        float bf_ = bx_c[0*U_+n], bi_ = bx_c[1*U_+n], bo_ = bx_c[2*U_+n], bg_ = bx_c[3*U_+n];
        #pragma unroll
        for (int mt = 0; mt < MT; ++mt) {
            #pragma unroll
            for (int v = 0; v < 8; ++v) {
                int rl = mt*16 + v + 8*lg;
                int r = rowBase + rl;
                size_t e = (size_t)r * U_ + n;
                float f = sigmoidf_(af[mt][v] + bf_);
                float i = sigmoidf_(ai[mt][v] + bi_);
                float o = sigmoidf_(ao[mt][v] + bo_);
                float g = tanhf(ag[mt][v] + bg_);
                bool gp = g_prev[e] > 0;
                float scn = i * g + (gp ? 0.f : f * s_c[e]);
                float hcn = o * tanhf(scn);
                out[O_SCN + e] = scn;
                out[O_HCN + e] = hcn;
                sB1[rl * LDSTR + n] = f2bf(hcn);   // h_c dead; stage h_c_n
            }
        }
    }
    __syncthreads();

    // ---- step 3: MLP_c -> h_c_p --------------------------------------------
    {                                          // GEMM1: N=128, wave w -> n-tile w
        int j = w;
        v8f acc[MT] = {};
        #pragma unroll
        for (int t = 0; t < 8; ++t) {
            v16bf b = bfrag(pk + OFF_WPC0, 8, j, t, lane);
            #pragma unroll
            for (int mt = 0; mt < MT; ++mt)
                acc[mt] = wmma_bf16(afrag(sB1, LDSTR, mt*16, t*32, lane), b, acc[mt]);
        }
        int n = (j << 4) + ln;
        float b0 = bpc0[n];
        #pragma unroll
        for (int mt = 0; mt < MT; ++mt)
            #pragma unroll
            for (int v = 0; v < 8; ++v)
                sA0[(mt*16 + v + 8*lg) * A0STR + n] = f2bf(leakyf_(acc[mt][v] + b0));
    }
    __syncthreads();
    if (w < 4) {                               // GEMM2: N=64, waves 0..3
        int j = w;
        v8f acc[MT] = {};
        #pragma unroll
        for (int t = 0; t < 4; ++t) {
            v16bf b = bfrag(pk + OFF_WPC1, 4, j, t, lane);
            #pragma unroll
            for (int mt = 0; mt < MT; ++mt)
                acc[mt] = wmma_bf16(afrag(sA0, A0STR, mt*16, t*32, lane), b, acc[mt]);
        }
        int n = (j << 4) + ln;
        float b1 = bpc1[n];
        #pragma unroll
        for (int mt = 0; mt < MT; ++mt)
            #pragma unroll
            for (int v = 0; v < 8; ++v)
                sA1[(mt*16 + v + 8*lg) * A1STR + n] = f2bf(leakyf_(acc[mt][v] + b1));
    }
    __syncthreads();
    if (tid < ROWS) {                          // GEMM3: N=1 dot product
        float acc = bfcc[0];
        for (int k = 0; k < 64; ++k) acc += bf2f(sA1[tid * A1STR + k]) * Wfcc[k];
        float p = sigmoidf_(acc);
        sP[tid] = p;
        out[O_HCP + rowBase + tid] = p;
    }
    __syncthreads();

    // ---- step 4: s_v_hat = h_v@W_svhat_v + gm * h_c_n@W_svhat_c ------------
    for (int jj = 0; jj < 2; ++jj) {
        int j = jj * 8 + w;
        v8f a1[MT] = {}, a2[MT] = {};
        #pragma unroll
        for (int t = 0; t < 8; ++t) {
            v16bf b1 = bfrag(pk + OFF_SVHAT_V, 8, j, t, lane);
            v16bf b2 = bfrag(pk + OFF_SVHAT_C, 8, j, t, lane);
            #pragma unroll
            for (int mt = 0; mt < MT; ++mt) {
                a1[mt] = wmma_bf16(afrag(sHv, LDSTR, mt*16, t*32, lane), b1, a1[mt]);
                a2[mt] = wmma_bf16(afrag(sB1, LDSTR, mt*16, t*32, lane), b2, a2[mt]);
            }
        }
        int n = (j << 4) + ln;
        #pragma unroll
        for (int mt = 0; mt < MT; ++mt) {
            #pragma unroll
            for (int v = 0; v < 8; ++v) {
                int rl = mt*16 + v + 8*lg;
                int r = rowBase + rl;
                bool gm = click[r] >= 0.5f;
                sT[rl * LDSTR + n] = f2bf(a1[mt][v] + (gm ? a2[mt][v] : 0.f));
            }
        }
    }
    __syncthreads();

    // ---- step 5: conversion-branch gates, s_v_n / h_v_n / g_new ------------
    for (int jj = 0; jj < 2; ++jj) {
        int j = jj * 8 + w;
        v8f af[MT] = {}, ai[MT] = {}, ao[MT] = {}, ag[MT] = {};
        #pragma unroll
        for (int t = 0; t < 8; ++t) {      // x @ Wx_v
            v16bf bF = bfrag(pk + OFF_WXV + 0*PKU, 8, j, t, lane);
            v16bf bI = bfrag(pk + OFF_WXV + 1*PKU, 8, j, t, lane);
            v16bf bO = bfrag(pk + OFF_WXV + 2*PKU, 8, j, t, lane);
            v16bf bG = bfrag(pk + OFF_WXV + 3*PKU, 8, j, t, lane);
            #pragma unroll
            for (int mt = 0; mt < MT; ++mt) {
                v16bf a = afrag(sX, LDSTR, mt*16, t*32, lane);
                af[mt] = wmma_bf16(a, bF, af[mt]);
                ai[mt] = wmma_bf16(a, bI, ai[mt]);
                ao[mt] = wmma_bf16(a, bO, ao[mt]);
                ag[mt] = wmma_bf16(a, bG, ag[mt]);
            }
        }
        #pragma unroll
        for (int t = 0; t < 8; ++t) {      // s_v_hat @ Wh_v
            v16bf bF = bfrag(pk + OFF_WHV + 0*PKU, 8, j, t, lane);
            v16bf bI = bfrag(pk + OFF_WHV + 1*PKU, 8, j, t, lane);
            v16bf bO = bfrag(pk + OFF_WHV + 2*PKU, 8, j, t, lane);
            v16bf bG = bfrag(pk + OFF_WHV + 3*PKU, 8, j, t, lane);
            #pragma unroll
            for (int mt = 0; mt < MT; ++mt) {
                v16bf a = afrag(sT, LDSTR, mt*16, t*32, lane);
                af[mt] = wmma_bf16(a, bF, af[mt]);
                ai[mt] = wmma_bf16(a, bI, ai[mt]);
                ao[mt] = wmma_bf16(a, bO, ao[mt]);
                ag[mt] = wmma_bf16(a, bG, ag[mt]);
            }
        }
        int n = (j << 4) + ln;
        float bf_ = bx_v[0*U_+n], bi_ = bx_v[1*U_+n], bo_ = bx_v[2*U_+n], bg_ = bx_v[3*U_+n];
        #pragma unroll
        for (int mt = 0; mt < MT; ++mt) {
            #pragma unroll
            for (int v = 0; v < 8; ++v) {
                int rl = mt*16 + v + 8*lg;
                int r = rowBase + rl;
                size_t e = (size_t)r * U_ + n;
                bool gm = click[r] >= 0.5f;
                float f = sigmoidf_(af[mt][v] + bf_);
                float i = sigmoidf_(ai[mt][v] + bi_);
                float o = sigmoidf_(ao[mt][v] + bo_);
                float g = tanhf(ag[mt][v] + bg_);
                float svn = gm ? (f * s_v[e] + i * g) : s_v[e];
                float hvn = gm ? (o * tanhf(svn)) : h_v[e];
                out[O_SVN + e] = svn;
                out[O_HVN + e] = hvn;
                out[O_GNEW + e] = gm ? 1.f : 0.f;
                sB1[rl * LDSTR + n] = f2bf(hvn);   // h_c_n dead; stage h_v_n
            }
        }
    }
    __syncthreads();

    // ---- step 6: MLP_v -> h_v_p = sigmoid(...) * h_c_p ---------------------
    {
        int j = w;
        v8f acc[MT] = {};
        #pragma unroll
        for (int t = 0; t < 8; ++t) {
            v16bf b = bfrag(pk + OFF_WPV0, 8, j, t, lane);
            #pragma unroll
            for (int mt = 0; mt < MT; ++mt)
                acc[mt] = wmma_bf16(afrag(sB1, LDSTR, mt*16, t*32, lane), b, acc[mt]);
        }
        int n = (j << 4) + ln;
        float b0 = bpv0[n];
        #pragma unroll
        for (int mt = 0; mt < MT; ++mt)
            #pragma unroll
            for (int v = 0; v < 8; ++v)
                sA0[(mt*16 + v + 8*lg) * A0STR + n] = f2bf(leakyf_(acc[mt][v] + b0));
    }
    __syncthreads();
    if (w < 4) {
        int j = w;
        v8f acc[MT] = {};
        #pragma unroll
        for (int t = 0; t < 4; ++t) {
            v16bf b = bfrag(pk + OFF_WPV1, 4, j, t, lane);
            #pragma unroll
            for (int mt = 0; mt < MT; ++mt)
                acc[mt] = wmma_bf16(afrag(sA0, A0STR, mt*16, t*32, lane), b, acc[mt]);
        }
        int n = (j << 4) + ln;
        float b1 = bpv1[n];
        #pragma unroll
        for (int mt = 0; mt < MT; ++mt)
            #pragma unroll
            for (int v = 0; v < 8; ++v)
                sA1[(mt*16 + v + 8*lg) * A1STR + n] = f2bf(leakyf_(acc[mt][v] + b1));
    }
    __syncthreads();
    if (tid < ROWS) {
        float acc = bfcv[0];
        for (int k = 0; k < 64; ++k) acc += bf2f(sA1[tid * A1STR + k]) * Wfcv[k];
        out[O_HVP + rowBase + tid] = sigmoidf_(acc) * sP[tid];
    }
}

// ---------- host entry -------------------------------------------------------
extern "C" void kernel_launch(void* const* d_in, const int* in_sizes, int n_in,
                              void* d_out, int out_size, void* d_ws, size_t ws_size,
                              hipStream_t stream) {
    const float* x         = (const float*)d_in[0];
    const float* click     = (const float*)d_in[1];
    const float* h_c       = (const float*)d_in[2];
    const float* h_v       = (const float*)d_in[3];
    const float* s_c       = (const float*)d_in[4];
    const float* s_v       = (const float*)d_in[5];
    const int*   g_prev    = (const int*)  d_in[6];
    const float* Wx_c      = (const float*)d_in[7];
    const float* bx_c      = (const float*)d_in[8];
    const float* Wh_c      = (const float*)d_in[9];
    const float* Wx_v      = (const float*)d_in[10];
    const float* bx_v      = (const float*)d_in[11];
    const float* Wh_v      = (const float*)d_in[12];
    const float* W_schat_c = (const float*)d_in[13];
    const float* W_schat_v = (const float*)d_in[14];
    const float* W_svhat_v = (const float*)d_in[15];
    const float* W_svhat_c = (const float*)d_in[16];
    const float* Wpc0      = (const float*)d_in[17];
    const float* bpc0      = (const float*)d_in[18];
    const float* Wpc1      = (const float*)d_in[19];
    const float* bpc1      = (const float*)d_in[20];
    const float* Wfcc      = (const float*)d_in[21];
    const float* bfcc      = (const float*)d_in[22];
    const float* Wpv0      = (const float*)d_in[23];
    const float* bpv0      = (const float*)d_in[24];
    const float* Wpv1      = (const float*)d_in[25];
    const float* bpv1      = (const float*)d_in[26];
    const float* Wfcv      = (const float*)d_in[27];
    const float* bfcv      = (const float*)d_in[28];

    __bf16* pk = (__bf16*)d_ws;
    auto packW = [&](const float* src, int K, int N, size_t off) {
        int th = (K >> 5) * (N >> 4) * 32;
        pack_bfrag<<<(th + 255) / 256, 256, 0, stream>>>(src, pk + off, K, N);
    };
    packW(W_schat_c, 256, 256, OFF_SCHAT_C);
    packW(W_schat_v, 256, 256, OFF_SCHAT_V);
    packW(W_svhat_v, 256, 256, OFF_SVHAT_V);
    packW(W_svhat_c, 256, 256, OFF_SVHAT_C);
    for (int g = 0; g < 4; ++g) {
        packW(Wx_c + (size_t)g * 65536, 256, 256, OFF_WXC + (size_t)g * PKU);
        packW(Wh_c + (size_t)g * 65536, 256, 256, OFF_WHC + (size_t)g * PKU);
        packW(Wx_v + (size_t)g * 65536, 256, 256, OFF_WXV + (size_t)g * PKU);
        packW(Wh_v + (size_t)g * 65536, 256, 256, OFF_WHV + (size_t)g * PKU);
    }
    packW(Wpc0, 256, 128, OFF_WPC0);
    packW(Wpv0, 256, 128, OFF_WPV0);
    packW(Wpc1, 128,  64, OFF_WPC1);
    packW(Wpv1, 128,  64, OFF_WPV1);

    heroes_fused<<<B_ / ROWS, THREADS, 0, stream>>>(
        x, click, h_c, h_v, s_c, s_v, g_prev,
        bx_c, bx_v, bpc0, bpc1, Wfcc, bfcc, bpv0, bpv1, Wfcv, bfcv,
        pk, (float*)d_out);
}
